// MultiHeadAttentionBase_37589553775300
// MI455X (gfx1250) — matrix operand order repaired and checked by reading, not compiled, not run on hardware
//
#include <hip/hip_runtime.h>

// ---------------------------------------------------------------------------
// MI455X (gfx1250) multi-head attention + head-merge, f16 WMMA flash-attention
// v4: masking folded into the WMMA C operand (S = Q*K^T + bias, bias=-1e30 on
// masked logits) -> no select chain in the hot loop. 64-wide KV tiles, base-2
// online softmax, all operands as aligned b128 loads.
// ---------------------------------------------------------------------------
typedef _Float16 h16;
typedef __attribute__((ext_vector_type(16))) _Float16 v16h;
typedef __attribute__((ext_vector_type(8)))  _Float16 v8h;
typedef __attribute__((ext_vector_type(8)))  float    v8f;

#define B_DIM 4
#define H_DIM 16
#define S_DIM 1024
#define P_DIM 64
#define D_DIM 1024   // H*P
#define QKV_N (B_DIM * H_DIM * S_DIM * P_DIM)  // 4194304 elements per tensor

// 1/sqrt(P) * log2(e): logits produced directly in base-2 domain
#define QSCALE   (0.125f * 1.44269504088896340736f)
#define MASK_NEG (-1.0e30f)

static __device__ __forceinline__ v8f wmma_f16(v16h a, v16h b, v8f c) {
  // (neg_a, A, neg_b, B, c_mod, C, reuse_a, reuse_b)
  return __builtin_amdgcn_wmma_f32_16x16x32_f16(false, a, false, b, (short)0, c,
                                                false, false);
}

// Assemble a 16x32 f16 A/B operand from two contiguous 8-half chunks.
// A-operand k-slots (ISA 7.12.2): slots 0..7 -> k=8g+0..7, slots 8..15 -> 16+8g+0..7
// This code fills A and B alike: slots 0..15 -> k=16g+0..15 (16 contiguous halves)
static __device__ __forceinline__ v16h cat8(v8h lo, v8h hi) {
  return __builtin_shufflevector(lo, hi, 0, 1, 2, 3, 4, 5, 6, 7, 8, 9, 10, 11,
                                 12, 13, 14, 15);
}

// ---------------------------------------------------------------------------
// Prep kernels (one-time)
// ---------------------------------------------------------------------------
__global__ void cvt_scale_kernel(const float* __restrict__ in,
                                 h16* __restrict__ out, float s, int n) {
  int i = blockIdx.x * blockDim.x + threadIdx.x;
  if (i < n) out[i] = (h16)(in[i] * s);
}

// v[BH, S, P] f32 -> vT[BH, P, S] f16 (coalesced writes; reads served by L2)
__global__ void transpose_v_kernel(const float* __restrict__ v,
                                   h16* __restrict__ vt) {
  int i = blockIdx.x * blockDim.x + threadIdx.x;  // index into vt
  int s  = i & (S_DIM - 1);
  int t  = i >> 10;
  int p  = t & (P_DIM - 1);
  int bh = t >> 6;
  vt[i] = (h16)v[((size_t)bh * S_DIM + s) * P_DIM + p];
}

// bias[b][t][s] = (posmask[t][s] | lenmask[b][s]) ? -1e30 : 0   (head-shared)
__global__ void build_bias_kernel(const unsigned char* __restrict__ pos,
                                  const unsigned char* __restrict__ len,
                                  float* __restrict__ bias) {
  int i = blockIdx.x * blockDim.x + threadIdx.x;  // over B*S*S
  int s = i & (S_DIM - 1);
  int t = (i >> 10) & (S_DIM - 1);
  int b = i >> 20;
  bias[i] =
      (pos[(size_t)t * S_DIM + s] | len[(size_t)b * S_DIM + s]) ? MASK_NEG
                                                                : 0.0f;
}

// ---------------------------------------------------------------------------
// Flash attention: one wave = one 16-row query tile of one (b,h), 64-wide KV
// tiles. Logits computed as Q*K^T + bias via the WMMA C operand. Online
// base-2 softmax in registers; P re-layout C->A via 2KB per-wave LDS.
// Output stored f16 in merged [B, T, H*P] layout.
// ---------------------------------------------------------------------------
__global__ __launch_bounds__(128) void flash_attn_kernel(
    const h16* __restrict__ qF, const h16* __restrict__ kF,
    const h16* __restrict__ vT, const float* __restrict__ bias,
    h16* __restrict__ dataF16) {
  __shared__ h16 ldsP[4][16 * 64];  // per-wave 16x64 f16 P tile

  const int wave = threadIdx.x >> 5;
  const int lane = threadIdx.x & 31;
  const int n16  = lane & 15;
  const int g    = lane >> 4;

  const int tile  = blockIdx.x * 4 + wave;   // 4096 tiles
  const int bh    = tile >> 6;
  const int qt    = tile & 63;
  const int b     = bh >> 4;
  const int hh    = bh & 15;
  const int qbase = qt * 16;

  // ---- Q tile as two A operands (k = 0..31, 32..63); scale pre-folded
  v16h qa[2];
  {
    const h16* qrow = qF + ((size_t)bh * S_DIM + qbase + n16) * P_DIM;
#pragma unroll
    for (int c = 0; c < 2; ++c)
      qa[c] = cat8(*(const v8h*)(qrow + 32 * c + 16 * g),
                   *(const v8h*)(qrow + 32 * c + 16 * g + 8));
  }

  const v8f vzero = {0.f, 0.f, 0.f, 0.f, 0.f, 0.f, 0.f, 0.f};
  float m_i[8], l_i[8];
  v8f o[4];
#pragma unroll
  for (int r = 0; r < 8; ++r) { m_i[r] = -__builtin_inff(); l_i[r] = 0.f; }
#pragma unroll
  for (int t4 = 0; t4 < 4; ++t4) o[t4] = vzero;

  // bias rows for this lane's C-tile slots: row qbase+r+8g, col = kv0+16j+n16
  const float* brow0 = bias + ((size_t)b * S_DIM + qbase + 8 * g) * S_DIM + n16;

  const int kv_end = qbase + 16;  // causal loop bound (matches triu(k=1) mask)

  for (int kv0 = 0; kv0 < kv_end; kv0 += 64) {
    // ---- S = Q * K^T + bias over a 16x64 logit tile (4 column sub-tiles)
    v8f s[4];
#pragma unroll
    for (int j = 0; j < 4; ++j) {
      const float* bp = brow0 + kv0 + 16 * j;
      v8f cinit;
#pragma unroll
      for (int r = 0; r < 8; ++r) cinit[r] = bp[(size_t)r * S_DIM];

      const h16* krow = kF + ((size_t)bh * S_DIM + kv0 + 16 * j + n16) * P_DIM;
      v16h kb0 = cat8(*(const v8h*)(krow + 16 * g),
                      *(const v8h*)(krow + 16 * g + 8));
      v16h kb1 = cat8(*(const v8h*)(krow + 32 + 16 * g),
                      *(const v8h*)(krow + 32 + 16 * g + 8));
      s[j] = wmma_f16(qa[0], kb0, cinit);
      s[j] = wmma_f16(qa[1], kb1, s[j]);
    }

    // ---- online base-2 softmax (16-lane butterfly inside each wave half)
    float sv[4][8];
#pragma unroll
    for (int r = 0; r < 8; ++r) {
      float mx = fmaxf(fmaxf(s[0][r], s[1][r]), fmaxf(s[2][r], s[3][r]));
#pragma unroll
      for (int d = 1; d < 16; d <<= 1) mx = fmaxf(mx, __shfl_xor(mx, d, 32));
      const float newm  = fmaxf(m_i[r], mx);
      const float scale = exp2f(m_i[r] - newm);
      float rs = 0.f;
#pragma unroll
      for (int j = 0; j < 4; ++j) {
        const float p = exp2f(s[j][r] - newm);
        sv[j][r] = p;
        rs += p;
      }
#pragma unroll
      for (int d = 1; d < 16; d <<= 1) rs += __shfl_xor(rs, d, 32);
      l_i[r] = l_i[r] * scale + rs;
      m_i[r] = newm;
#pragma unroll
      for (int t4 = 0; t4 < 4; ++t4) o[t4][r] *= scale;
    }

    // ---- P: C-layout -> A-layout via per-wave LDS (b128 read-back)
#pragma unroll
    for (int j = 0; j < 4; ++j)
#pragma unroll
      for (int r = 0; r < 8; ++r)
        ldsP[wave][(r + 8 * g) * 64 + 16 * j + n16] = (h16)sv[j][r];
    v16h pa[2];
#pragma unroll
    for (int c = 0; c < 2; ++c)
      pa[c] = cat8(*(const v8h*)(&ldsP[wave][n16 * 64 + 32 * c + 16 * g]),
                   *(const v8h*)(&ldsP[wave][n16 * 64 + 32 * c + 16 * g + 8]));

    // ---- O += P * V: vT gives 16 contiguous halves per lane (b128 loads)
#pragma unroll
    for (int t4 = 0; t4 < 4; ++t4) {
      const h16* vcol =
          vT + ((size_t)bh * P_DIM + 16 * t4 + n16) * S_DIM + kv0 + 16 * g;
      v16h vb0 = cat8(*(const v8h*)(vcol), *(const v8h*)(vcol + 8));
      v16h vb1 = cat8(*(const v8h*)(vcol + 32), *(const v8h*)(vcol + 40));
      o[t4] = wmma_f16(pa[0], vb0, o[t4]);
      o[t4] = wmma_f16(pa[1], vb1, o[t4]);
    }
  }

  // ---- normalize + store f16 in merged [B, T, H*P] layout
#pragma unroll
  for (int t4 = 0; t4 < 4; ++t4) {
#pragma unroll
    for (int r = 0; r < 8; ++r) {
      const int trow = qbase + r + 8 * g;
      dataF16[((size_t)b * S_DIM + trow) * D_DIM + hh * P_DIM + 16 * t4 + n16] =
          (h16)(o[t4][r] / l_i[r]);
    }
  }
}

// ---------------------------------------------------------------------------
// Merge GEMM: out[M,N] = data[M,K] * w[N,K]^T, M=4096, N=K=1024.
// One wave = 16M x 64N strip (4 accumulators): A loaded once per k-step,
// 4 WMMAs per k-step. 8 waves/block across N.
// ---------------------------------------------------------------------------
__global__ __launch_bounds__(256) void merge_gemm_kernel(
    const h16* __restrict__ dataF16, const h16* __restrict__ wF16,
    float* __restrict__ out) {
  const int wave = threadIdx.x >> 5;
  const int lane = threadIdx.x & 31;
  const int n16  = lane & 15;
  const int g    = lane >> 4;
  const int mbase = blockIdx.y * 16;
  const int nbase = (blockIdx.x * 8 + wave) * 64;  // 64-wide N strip

  const h16* arow = dataF16 + (size_t)(mbase + n16) * D_DIM;

  const v8f vzero = {0.f, 0.f, 0.f, 0.f, 0.f, 0.f, 0.f, 0.f};
  v8f acc[4];
#pragma unroll
  for (int t = 0; t < 4; ++t) acc[t] = vzero;

#pragma unroll 4
  for (int kc = 0; kc < D_DIM; kc += 32) {
    v16h a = cat8(*(const v8h*)(arow + kc + 16 * g),
                  *(const v8h*)(arow + kc + 16 * g + 8));
#pragma unroll
    for (int t = 0; t < 4; ++t) {
      const h16* brow = wF16 + (size_t)(nbase + 16 * t + n16) * D_DIM + kc;
      v16h bm = cat8(*(const v8h*)(brow + 16 * g),
                     *(const v8h*)(brow + 16 * g + 8));
      acc[t] = wmma_f16(a, bm, acc[t]);
    }
  }

#pragma unroll
  for (int t = 0; t < 4; ++t)
#pragma unroll
    for (int r = 0; r < 8; ++r)
      out[(size_t)(mbase + r + 8 * g) * D_DIM + nbase + 16 * t + n16] =
          acc[t][r];
}

// ---------------------------------------------------------------------------
extern "C" void kernel_launch(void* const* d_in, const int* in_sizes, int n_in,
                              void* d_out, int out_size, void* d_ws,
                              size_t ws_size, hipStream_t stream) {
  (void)in_sizes; (void)n_in; (void)out_size; (void)ws_size;
  const float* q = (const float*)d_in[0];
  const float* k = (const float*)d_in[1];
  const float* v = (const float*)d_in[2];
  const float* w = (const float*)d_in[3];
  const unsigned char* posmask = (const unsigned char*)d_in[4];
  const unsigned char* lenmask = (const unsigned char*)d_in[5];
  float* out = (float*)d_out;

  // workspace layout: qF | kF | vT | data | wF (halves), then bias (f32)
  h16* qF16    = (h16*)d_ws;
  h16* kF16    = qF16 + (size_t)QKV_N;
  h16* vTF16   = kF16 + (size_t)QKV_N;
  h16* dataF16 = vTF16 + (size_t)QKV_N;
  h16* wF16    = dataF16 + (size_t)B_DIM * S_DIM * D_DIM;
  float* bias  = (float*)(wF16 + (size_t)D_DIM * D_DIM);

  const int nq = QKV_N;
  cvt_scale_kernel<<<(nq + 255) / 256, 256, 0, stream>>>(q, qF16, QSCALE, nq);
  cvt_scale_kernel<<<(nq + 255) / 256, 256, 0, stream>>>(k, kF16, 1.0f, nq);
  cvt_scale_kernel<<<(D_DIM * D_DIM + 255) / 256, 256, 0, stream>>>(
      w, wF16, 1.0f, D_DIM * D_DIM);
  transpose_v_kernel<<<(nq + 255) / 256, 256, 0, stream>>>(v, vTF16);
  const int nb = B_DIM * S_DIM * S_DIM;
  build_bias_kernel<<<(nb + 255) / 256, 256, 0, stream>>>(posmask, lenmask,
                                                          bias);

  // 64 (b,h) * 64 q-tiles = 4096 wave-tiles, 4 waves/block
  flash_attn_kernel<<<(B_DIM * H_DIM * (S_DIM / 16)) / 4, 128, 0, stream>>>(
      qF16, kF16, vTF16, bias, dataF16);

  // N strips: 1024/64 = 16 -> grid.x = 2 (8 waves/block); M tiles: 256
  dim3 ggrid(2, 256);
  merge_gemm_kernel<<<ggrid, 256, 0, stream>>>(dataF16, wF16, out);
}
